// GNNTransformerDetectorV12_89781996356269
// MI455X (gfx1250) — matrix operand order repaired and checked by reading, not compiled, not run on hardware
//
#include <hip/hip_runtime.h>
#include <hip/hip_bf16.h>

// ---------------------------------------------------------------------------
// GNNTransformerDetector for MI455X (gfx1250): WMMA f16 GEMM core, wave32.
// Round 3: B-fragment preload (load-clause + back-to-back WMMA per k-step),
// TDM tensor_load_to_lds staging in the transformer kernel, async global->LDS
// staging in the feature kernel.
// ---------------------------------------------------------------------------

#define Bq 64
#define Lq 256
#define Kq 16
#define Nq 8
#define Hh 128
#define Pn (Bq * Lq * Kq)   // 262144
#define XLD 96              // padded fusion-input width (70 -> 96)

typedef __attribute__((ext_vector_type(16))) _Float16 v16h;
typedef __attribute__((ext_vector_type(8)))  _Float16 v8h;
typedef __attribute__((ext_vector_type(8)))  float    v8f;
typedef __attribute__((ext_vector_type(4)))  unsigned u32x4;
typedef __attribute__((ext_vector_type(8)))  int      i32x8;
typedef __attribute__((ext_vector_type(4)))  int      i32x4;

__device__ __forceinline__ v8f wmma16(v16h a, v16h b, v8f c) {
  // v_wmma_f32_16x16x32_f16: D = A(16x32) x B(32x16) + C(16x16 f32)
  return __builtin_amdgcn_wmma_f32_16x16x32_f16(false, a, false, b, (short)0, c,
                                                false, false);
}

__device__ __forceinline__ v8f vzero8() {
  v8f z = {0.f, 0.f, 0.f, 0.f, 0.f, 0.f, 0.f, 0.f};
  return z;
}

// A fragment (16x32 f16, row-major source, ld in elements).
__device__ __forceinline__ v16h load_a16(const _Float16* tile, int ld, int lane) {
  const int r  = lane & 15;
  const int c0 = (lane & 16) ? 16 : 0;
  const _Float16* p = tile + (size_t)r * ld + c0;
  v8h lo = *(const v8h*)p;
  v8h hi = *(const v8h*)(p + 8);
  v16h o;
#pragma unroll
  for (int e = 0; e < 8; ++e) { o[e] = lo[e]; o[e + 8] = hi[e]; }
  return o;
}

// Pre-packed B fragment: one contiguous 32B load per lane.
__device__ __forceinline__ v16h load_bp(const _Float16* __restrict__ Wp,
                                        int frag, int lane) {
  return *(const v16h*)(Wp + (((size_t)frag * 32 + lane) << 4));
}

// Pack f32 weights W[out][in] into WMMA B-fragment order.
__global__ __launch_bounds__(256) void k_pack(
    const float* __restrict__ W, int ldw, int kmax, int ktiles, int ntiles,
    _Float16* __restrict__ Wp) {
  const int idx = blockIdx.x * 256 + threadIdx.x;
  const int total = ktiles * ntiles * 32;
  if (idx >= total) return;
  const int lane = idx & 31;
  const int frag = idx >> 5;
  const int nt = frag % ntiles;
  const int kt = frag / ntiles;
  const int nrow = nt * 16 + (lane & 15);
  const int koff = kt * 32 + ((lane & 16) ? 16 : 0);
  _Float16* dst = Wp + (size_t)idx * 16;
#pragma unroll
  for (int e = 0; e < 16; ++e) {
    const int k = koff + e;
    dst[e] = (_Float16)((k < kmax) ? W[(size_t)nrow * ldw + k] : 0.f);
  }
}

// ---------------------------------------------------------------------------
// Kernel 0: MRC  s_mrc[b,k] = sum_{l,n} conj(H) * y ; mrc_embed = MLP(2->64->128)
// ---------------------------------------------------------------------------
__global__ __launch_bounds__(256) void k_mrc(
    const float* __restrict__ Hv, const float* __restrict__ yv,
    const float* __restrict__ W0, const float* __restrict__ B0,
    const float* __restrict__ W1, const float* __restrict__ B1,
    float* __restrict__ mrc) {
  __shared__ float sre[16][16], sim[16][16];
  const int b = blockIdx.x, tid = threadIdx.x;
  const int k = tid & 15, part = tid >> 4;
  float re = 0.f, im = 0.f;
  for (int ln = part; ln < Lq * Nq; ln += 16) {
    const int l = ln / Nq, n = ln % Nq;
    const size_t hidx = ((((size_t)b * Lq + l) * Nq + n) * Kq + k) * 2;
    const size_t yidx = (((size_t)b * Lq + l) * Nq + n) * 2;
    const float hr = Hv[hidx], hi = Hv[hidx + 1];
    const float yr = yv[yidx], yi = yv[yidx + 1];
    re += hr * yr + hi * yi;
    im += hr * yi - hi * yr;
  }
  sre[k][part] = re; sim[k][part] = im;
  __syncthreads();
  if (part == 0) {
    float R = 0.f, I = 0.f;
#pragma unroll
    for (int j = 0; j < 16; ++j) { R += sre[k][j]; I += sim[k][j]; }
    float hid[64];
#pragma unroll
    for (int u = 0; u < 64; ++u)
      hid[u] = fmaxf(W0[u * 2] * R + W0[u * 2 + 1] * I + B0[u], 0.f);
    float* out = mrc + ((size_t)b * Kq + k) * Hh;
    for (int o = 0; o < Hh; ++o) {
      float a = B1[o];
#pragma unroll
      for (int u = 0; u < 64; ++u) a += hid[u] * W1[o * 64 + u];
      out[o] = a;
    }
  }
}

// ---------------------------------------------------------------------------
// Kernel 1: per-(b,l) features -> X[P][96] f16.  H block staged into LDS with
// CDNA5 async global->LDS copies (ASYNCcnt-tracked).
// ---------------------------------------------------------------------------
__global__ __launch_bounds__(128) void k_feat(
    const float* __restrict__ sh, const float* __restrict__ Hv,
    const float* __restrict__ bw, const float* __restrict__ snr,
    const float* __restrict__ dW0, const float* __restrict__ dB0,
    const float* __restrict__ dW1, const float* __restrict__ dB1,
    const float* __restrict__ cW0, const float* __restrict__ cB0,
    const float* __restrict__ cW1, const float* __restrict__ cB1,
    _Float16* __restrict__ X) {
  __shared__ float hstage[Nq * Kq * 2];  // 256 floats, [n][k][c]
  __shared__ float chs[16][16];
  __shared__ float hpp[16][8];
  __shared__ float hpow[16];
  __shared__ float dmh[16][32];
  __shared__ float chh[16][32];
  __shared__ float tot_s;
  const int bl = blockIdx.x, tid = threadIdx.x;
  {
    const float* gsrc = Hv + (size_t)bl * Nq * Kq * 2 + tid * 2;
    const unsigned loff = (unsigned)(uintptr_t)(&hstage[tid * 2]);
    asm volatile("global_load_async_to_lds_b64 %0, %1, off"
                 :: "v"(loff), "v"(gsrc) : "memory");
    asm volatile("s_wait_asynccnt 0" ::: "memory");
  }
  __syncthreads();
  {
    const int n = tid >> 4, k = tid & 15;
    const float c0 = hstage[(n * Kq + k) * 2];
    const float c1 = hstage[(n * Kq + k) * 2 + 1];
    chs[k][2 * n] = c0; chs[k][2 * n + 1] = c1;
    hpp[k][n] = c0 * c0 + c1 * c1;
  }
  __syncthreads();
  if (tid < 16) {
    float s = 0.f;
#pragma unroll
    for (int j = 0; j < 8; ++j) s += hpp[tid][j];
    hpow[tid] = s;
  }
  __syncthreads();
  if (tid == 0) {
    float t = 0.f;
#pragma unroll
    for (int j = 0; j < 16; ++j) t += hpow[j];
    tot_s = t;
  }
  {
    const int k = tid >> 3, j = tid & 7;
    const size_t sidx = ((size_t)bl * Kq + k) * 2;
    const float s0 = sh[sidx], s1 = sh[sidx + 1];
#pragma unroll
    for (int u = j * 4; u < j * 4 + 4; ++u)
      dmh[k][u] = fmaxf(dW0[u * 2] * s0 + dW0[u * 2 + 1] * s1 + dB0[u], 0.f);
  }
  __syncthreads();
  {
    const int k = tid >> 3, j = tid & 7;
#pragma unroll
    for (int u = j * 4; u < j * 4 + 4; ++u) {
      float a = cB0[u];
#pragma unroll
      for (int m = 0; m < 16; ++m) a += cW0[u * 16 + m] * chs[k][m];
      chh[k][u] = fmaxf(a, 0.f);
    }
  }
  __syncthreads();
  {
    const int k = tid >> 3, j = tid & 7;
    const size_t p = (size_t)bl * Kq + k;
#pragma unroll
    for (int u = j * 4; u < j * 4 + 4; ++u) {
      float a = dB1[u];
#pragma unroll
      for (int m = 0; m < 32; ++m) a += dW1[u * 32 + m] * dmh[k][m];
      X[p * XLD + u] = (_Float16)a;
      float a2 = cB1[u];
#pragma unroll
      for (int m = 0; m < 32; ++m) a2 += cW1[u * 32 + m] * chh[k][m];
      X[p * XLD + 32 + u] = (_Float16)a2;
    }
  }
  if (tid < 16) {
    const int k = tid;
    const size_t p = (size_t)bl * Kq + k;
    X[p * XLD + 64] = (_Float16)bw[p * 3 + 0];
    X[p * XLD + 65] = (_Float16)bw[p * 3 + 1];
    X[p * XLD + 66] = (_Float16)bw[p * 3 + 2];
    X[p * XLD + 67] = (_Float16)log1pf(hpow[k]);
    X[p * XLD + 68] = (_Float16)log1pf(tot_s - hpow[k]);
    X[p * XLD + 69] = (_Float16)snr[p];
#pragma unroll
    for (int c = 70; c < XLD; ++c) X[p * XLD + c] = (_Float16)0.f;
  }
}

// ---------------------------------------------------------------------------
// Kernel 2: fusion MLP (WMMA)  node = W2 * relu(W1 * x + b1) + b2
// B fragments preloaded per k-step -> one load clause + 8 back-to-back WMMAs.
// ---------------------------------------------------------------------------
__global__ __launch_bounds__(128) void k_fusion(
    const _Float16* __restrict__ X,
    const _Float16* __restrict__ W1p, const float* __restrict__ B1,
    const _Float16* __restrict__ W2p, const float* __restrict__ B2,
    _Float16* __restrict__ node) {
  __shared__ _Float16 hid[4][16][Hh];
  const int tid = threadIdx.x, lane = tid & 31, w = tid >> 5;
  const int rowBase = blockIdx.x * 64 + w * 16;
  const int n = lane & 15, hi = lane >> 4;
  v8f acc[8];
#pragma unroll
  for (int t = 0; t < 8; ++t) acc[t] = vzero8();
  for (int kt = 0; kt < 3; ++kt) {
    v16h a = load_a16(X + (size_t)rowBase * XLD + kt * 32, XLD, lane);
    v16h bf[8];
#pragma unroll
    for (int nt = 0; nt < 8; ++nt) bf[nt] = load_bp(W1p, kt * 8 + nt, lane);
#pragma unroll
    for (int nt = 0; nt < 8; ++nt) acc[nt] = wmma16(a, bf[nt], acc[nt]);
  }
#pragma unroll
  for (int nt = 0; nt < 8; ++nt) {
    const float bias = B1[nt * 16 + n];
#pragma unroll
    for (int i = 0; i < 8; ++i) {
      float v = acc[nt][i] + bias;
      hid[w][i + 8 * hi][nt * 16 + n] = (_Float16)(v > 0.f ? v : 0.f);
    }
    acc[nt] = vzero8();
  }
  __syncthreads();
  for (int kt = 0; kt < 4; ++kt) {
    v16h a = load_a16(&hid[w][0][0] + kt * 32, Hh, lane);
    v16h bf[8];
#pragma unroll
    for (int nt = 0; nt < 8; ++nt) bf[nt] = load_bp(W2p, kt * 8 + nt, lane);
#pragma unroll
    for (int nt = 0; nt < 8; ++nt) acc[nt] = wmma16(a, bf[nt], acc[nt]);
  }
#pragma unroll
  for (int nt = 0; nt < 8; ++nt) {
    const float bias = B2[nt * 16 + n];
#pragma unroll
    for (int i = 0; i < 8; ++i)
      node[(size_t)(rowBase + i + 8 * hi) * Hh + nt * 16 + n] =
          (_Float16)(acc[nt][i] + bias);
  }
}

// ---------------------------------------------------------------------------
// Kernel 3: batchnorm stats (sum / sumsq per channel, atomics)
// ---------------------------------------------------------------------------
__global__ __launch_bounds__(256) void k_bnstat(
    const _Float16* __restrict__ node, float* __restrict__ bsum,
    float* __restrict__ bss) {
  const int tid = threadIdx.x, ch = tid & 127, sub = tid >> 7;
  float s = 0.f, ss = 0.f;
  const size_t step = (size_t)gridDim.x * 2;
  for (size_t r = (size_t)blockIdx.x * 2 + sub; r < Pn; r += step) {
    __builtin_prefetch(node + (r + step) * Hh + ch, 0, 0);
    const float v = (float)node[r * Hh + ch];
    s += v; ss += v * v;
  }
  atomicAdd(&bsum[ch], s);
  atomicAdd(&bss[ch], ss);
}

// ---------------------------------------------------------------------------
// Kernel 4: batchnorm apply -> h (f16)
// ---------------------------------------------------------------------------
__global__ __launch_bounds__(256) void k_bnapply(
    const _Float16* __restrict__ node, const float* __restrict__ bsum,
    const float* __restrict__ bss, const float* __restrict__ g,
    const float* __restrict__ bb, _Float16* __restrict__ h) {
  const size_t tot = (size_t)Pn * Hh;
  const float invP = 1.0f / (float)Pn;
  for (size_t i = (size_t)blockIdx.x * 256 + threadIdx.x; i < tot;
       i += (size_t)gridDim.x * 256) {
    const int ch = (int)(i & 127);
    const float m = bsum[ch] * invP;
    const float var = bss[ch] * invP - m * m;
    const float v = (float)node[i];
    h[i] = (_Float16)((v - m) * rsqrtf(var + 1e-5f) * g[ch] + bb[ch]);
  }
}

// ---------------------------------------------------------------------------
// Kernel 5: GNN msg MLP (WMMA) + mean over L via global f32 atomics
// ---------------------------------------------------------------------------
__global__ __launch_bounds__(128) void k_msg(
    const _Float16* __restrict__ h,
    const _Float16* __restrict__ W1p, const float* __restrict__ B1,
    const _Float16* __restrict__ W2p, const float* __restrict__ B2,
    float* __restrict__ msum) {
  __shared__ _Float16 hid[4][16][Hh];
  const int tid = threadIdx.x, lane = tid & 31, w = tid >> 5;
  const int rowBase = blockIdx.x * 64 + w * 16;
  const int n = lane & 15, hi = lane >> 4;
  const int bIdx = (rowBase >> 4) / Lq;
  v8f acc[8];
#pragma unroll
  for (int t = 0; t < 8; ++t) acc[t] = vzero8();
  for (int kt = 0; kt < 4; ++kt) {
    v16h a = load_a16(h + (size_t)rowBase * Hh + kt * 32, Hh, lane);
    v16h bf[8];
#pragma unroll
    for (int nt = 0; nt < 8; ++nt) bf[nt] = load_bp(W1p, kt * 8 + nt, lane);
#pragma unroll
    for (int nt = 0; nt < 8; ++nt) acc[nt] = wmma16(a, bf[nt], acc[nt]);
  }
#pragma unroll
  for (int nt = 0; nt < 8; ++nt) {
    const float bias = B1[nt * 16 + n];
#pragma unroll
    for (int i = 0; i < 8; ++i) {
      float v = acc[nt][i] + bias;
      hid[w][i + 8 * hi][nt * 16 + n] = (_Float16)(v > 0.f ? v : 0.f);
    }
    acc[nt] = vzero8();
  }
  __syncthreads();
  for (int kt = 0; kt < 4; ++kt) {
    v16h a = load_a16(&hid[w][0][0] + kt * 32, Hh, lane);
    v16h bf[8];
#pragma unroll
    for (int nt = 0; nt < 8; ++nt) bf[nt] = load_bp(W2p, kt * 8 + nt, lane);
#pragma unroll
    for (int nt = 0; nt < 8; ++nt) acc[nt] = wmma16(a, bf[nt], acc[nt]);
  }
#pragma unroll
  for (int nt = 0; nt < 8; ++nt) {
    const float bias = B2[nt * 16 + n];
#pragma unroll
    for (int i = 0; i < 8; ++i) {
      const int k = i + 8 * hi;
      atomicAdd(&msum[((size_t)bIdx * Kq + k) * Hh + nt * 16 + n],
                acc[nt][i] + bias);
    }
  }
}

// ---------------------------------------------------------------------------
// Kernel 6: GNN upd MLP (K=256 concat [h, msg_mean]) + residual + LayerNorm
// ---------------------------------------------------------------------------
__global__ __launch_bounds__(128) void k_upd(
    _Float16* __restrict__ h, const float* __restrict__ msum,
    const _Float16* __restrict__ W1p, const float* __restrict__ B1,
    const _Float16* __restrict__ W2p, const float* __restrict__ B2,
    const float* __restrict__ lng, const float* __restrict__ lnb) {
  __shared__ _Float16 hid[4][16][Hh];
  __shared__ float lnbuf[4][16][Hh];
  const int tid = threadIdx.x, lane = tid & 31, w = tid >> 5;
  const int rowBase = blockIdx.x * 64 + w * 16;
  const int n = lane & 15, hi = lane >> 4;
  const int bIdx = (rowBase >> 4) / Lq;
  const float invL = 1.0f / (float)Lq;
  v8f acc[8];
#pragma unroll
  for (int t = 0; t < 8; ++t) acc[t] = vzero8();
  for (int kt = 0; kt < 8; ++kt) {
    v16h a;
    if (kt < 4) {
      a = load_a16(h + (size_t)rowBase * Hh + kt * 32, Hh, lane);
    } else {
      const int r = lane & 15;
      const int c0 = (kt - 4) * 32 + ((lane & 16) ? 16 : 0);
      const float* p = msum + ((size_t)bIdx * Kq + r) * Hh + c0;
#pragma unroll
      for (int e = 0; e < 8; ++e) {
        a[e]     = (_Float16)(p[e] * invL);
        a[e + 8] = (_Float16)(p[e + 8] * invL);
      }
    }
    v16h bf[8];
#pragma unroll
    for (int nt = 0; nt < 8; ++nt) bf[nt] = load_bp(W1p, kt * 8 + nt, lane);
#pragma unroll
    for (int nt = 0; nt < 8; ++nt) acc[nt] = wmma16(a, bf[nt], acc[nt]);
  }
#pragma unroll
  for (int nt = 0; nt < 8; ++nt) {
    const float bias = B1[nt * 16 + n];
#pragma unroll
    for (int i = 0; i < 8; ++i) {
      float v = acc[nt][i] + bias;
      hid[w][i + 8 * hi][nt * 16 + n] = (_Float16)(v > 0.f ? v : 0.f);
    }
    acc[nt] = vzero8();
  }
  __syncthreads();
  for (int kt = 0; kt < 4; ++kt) {
    v16h a = load_a16(&hid[w][0][0] + kt * 32, Hh, lane);
    v16h bf[8];
#pragma unroll
    for (int nt = 0; nt < 8; ++nt) bf[nt] = load_bp(W2p, kt * 8 + nt, lane);
#pragma unroll
    for (int nt = 0; nt < 8; ++nt) acc[nt] = wmma16(a, bf[nt], acc[nt]);
  }
#pragma unroll
  for (int nt = 0; nt < 8; ++nt) {
    const float bias = B2[nt * 16 + n];
#pragma unroll
    for (int i = 0; i < 8; ++i)
      lnbuf[w][i + 8 * hi][nt * 16 + n] = acc[nt][i] + bias;
  }
  for (int r = 0; r < 16; ++r) {
    const size_t row = (size_t)rowBase + r;
    const _Float16* hr = h + row * Hh;
    float v[4], s = 0.f, ss = 0.f;
#pragma unroll
    for (int j = 0; j < 4; ++j) {
      const int c = lane * 4 + j;
      const float x = lnbuf[w][r][c] + (float)hr[c];
      v[j] = x; s += x; ss += x * x;
    }
#pragma unroll
    for (int m = 16; m >= 1; m >>= 1) {
      s += __shfl_xor(s, m, 32);
      ss += __shfl_xor(ss, m, 32);
    }
    const float mean = s * (1.0f / Hh);
    const float var = ss * (1.0f / Hh) - mean * mean;
    const float rstd = rsqrtf(var + 1e-5f);
#pragma unroll
    for (int j = 0; j < 4; ++j) {
      const int c = lane * 4 + j;
      h[row * Hh + c] = (_Float16)((v[j] - mean) * rstd * lng[c] + lnb[c]);
    }
  }
}

// ---------------------------------------------------------------------------
// Kernel 7a: attention-pool scores via WMMA (128->64 relu) + wave dot (64->1)
// ---------------------------------------------------------------------------
__global__ __launch_bounds__(128) void k_score(
    const _Float16* __restrict__ h, const _Float16* __restrict__ W0p,
    const float* __restrict__ B0, const float* __restrict__ W1,
    const float* __restrict__ B1, float* __restrict__ scores) {
  __shared__ _Float16 hidb[4][16][64];
  const int tid = threadIdx.x, lane = tid & 31, w = tid >> 5;
  const int rowBase = blockIdx.x * 64 + w * 16;
  const int n = lane & 15, hi = lane >> 4;
  v8f acc[4];
#pragma unroll
  for (int t = 0; t < 4; ++t) acc[t] = vzero8();
  for (int kt = 0; kt < 4; ++kt) {
    v16h a = load_a16(h + (size_t)rowBase * Hh + kt * 32, Hh, lane);
    v16h bf[4];
#pragma unroll
    for (int nt = 0; nt < 4; ++nt) bf[nt] = load_bp(W0p, kt * 4 + nt, lane);
#pragma unroll
    for (int nt = 0; nt < 4; ++nt) acc[nt] = wmma16(a, bf[nt], acc[nt]);
  }
#pragma unroll
  for (int nt = 0; nt < 4; ++nt) {
    const float bias = B0[nt * 16 + n];
#pragma unroll
    for (int i = 0; i < 8; ++i) {
      float v = acc[nt][i] + bias;
      hidb[w][i + 8 * hi][nt * 16 + n] = (_Float16)(v > 0.f ? v : 0.f);
    }
  }
  const float w1a = W1[lane * 2], w1b = W1[lane * 2 + 1];
  const float b1 = B1[0];
  for (int r = 0; r < 16; ++r) {
    float v = (float)hidb[w][r][lane * 2] * w1a +
              (float)hidb[w][r][lane * 2 + 1] * w1b;
#pragma unroll
    for (int m = 16; m >= 1; m >>= 1) v += __shfl_xor(v, m, 32);
    if (lane == 0) scores[rowBase + r] = v + b1;
  }
}

// ---------------------------------------------------------------------------
// Kernel 7b: softmax over L + weighted sum + mrc_embed -> xk[B,K,H]
// ---------------------------------------------------------------------------
__global__ __launch_bounds__(128) void k_pool(
    const _Float16* __restrict__ h, const float* __restrict__ scores,
    const float* __restrict__ mrc, float* __restrict__ xk) {
  __shared__ float sc[Lq];
  __shared__ float red[128];
  const int bk = blockIdx.x, b = bk >> 4, k = bk & 15;
  const int tid = threadIdx.x;
  for (int l = tid; l < Lq; l += 128)
    sc[l] = scores[(size_t)(b * Lq + l) * Kq + k];
  __syncthreads();
  float m = -1e30f;
  for (int l = tid; l < Lq; l += 128) m = fmaxf(m, sc[l]);
  red[tid] = m;
  __syncthreads();
  for (int s = 64; s > 0; s >>= 1) {
    if (tid < s) red[tid] = fmaxf(red[tid], red[tid + s]);
    __syncthreads();
  }
  const float mx = red[0];
  __syncthreads();
  float s0 = 0.f;
  for (int l = tid; l < Lq; l += 128) {
    const float e = __expf(sc[l] - mx);
    sc[l] = e; s0 += e;
  }
  red[tid] = s0;
  __syncthreads();
  for (int s = 64; s > 0; s >>= 1) {
    if (tid < s) red[tid] += red[tid + s];
    __syncthreads();
  }
  const float inv = 1.0f / red[0];
  float acc = 0.f;
  for (int l = 0; l < Lq; ++l) {
    const size_t idx = ((size_t)(b * Lq + l) * Kq + k) * Hh + tid;
    __builtin_prefetch(h + idx + (size_t)Kq * Hh, 0, 0);
    acc += (float)h[idx] * sc[l];
  }
  const size_t o = ((size_t)b * Kq + k) * Hh + tid;
  xk[o] = acc * inv + mrc[o];
}

// ---------------------------------------------------------------------------
// Kernel 8: 2-layer pre-norm transformer over K=16 tokens (per batch, in LDS).
// xk tile staged into LDS via the Tensor Data Mover (TENSORcnt-tracked).
// ---------------------------------------------------------------------------
struct TFW {
  const _Float16 *ipP, *opP, *f1P, *f2P;  // pre-packed f16 weights
  const float *ipB, *opB, *l1g, *l1b, *l2g, *l2b, *f1B, *f2B;
};

__global__ __launch_bounds__(128) void k_tf(float* __restrict__ xk, TFW t0, TFW t1) {
  __shared__ float    xkb[16][Hh];
  __shared__ _Float16 ab[16][Hh];
  __shared__ _Float16 qkvb[16][384];
  __shared__ float    scb[4][16][16];
  __shared__ _Float16 pb[4][16][32];
  __shared__ _Float16 ob[16][Hh];
  __shared__ _Float16 hb[16][256];
  const int b = blockIdx.x, tid = threadIdx.x, lane = tid & 31, w = tid >> 5;
  const int n = lane & 15, hi = lane >> 4;
  float* xg = xk + (size_t)b * Kq * Hh;
  // ---- stage xk (16x128 f32 = 8KB contiguous) into LDS via TDM ----
#if defined(__has_builtin) && __has_builtin(__builtin_amdgcn_tensor_load_to_lds)
  if (w == 0) {
    const unsigned lds_addr = (unsigned)(uintptr_t)(&xkb[0][0]);
    const unsigned long long ga = (unsigned long long)(uintptr_t)xg;
    u32x4 g0;
    g0[0] = 1u;                                   // count=1, user mode
    g0[1] = lds_addr;                             // D#.lds_addr
    g0[2] = (unsigned)(ga & 0xFFFFFFFFu);         // global_addr[31:0]
    g0[3] = (unsigned)((ga >> 32) & 0x01FFFFFFu) | (2u << 30);  // [56:32]|type=2
    i32x8 g1;
    g1[0] = 0x00020000;                 // data_size=4B; no multicast/pad/iter
    g1[1] = (int)((2048u & 0xFFFFu) << 16);       // tensor_dim0 = 2048 (lo16)
    g1[2] = (int)((2048u >> 16) | (1u << 16));    // tensor_dim0 hi | tensor_dim1=1
    g1[3] = (int)(2048u << 16);                   // tile_dim0 = 2048
    g1[4] = 0;                                    // tile_dim1/2 unused
    g1[5] = 2048;                                 // tensor_dim0_stride
    g1[6] = 0; g1[7] = 0;
    i32x4 gz = {0, 0, 0, 0};
#if __clang_major__ >= 23
    i32x8 gz8 = {0, 0, 0, 0, 0, 0, 0, 0};
    __builtin_amdgcn_tensor_load_to_lds(g0, g1, gz, gz, gz8, 0);
#else
    __builtin_amdgcn_tensor_load_to_lds(g0, g1, gz, gz, 0);
#endif
    __builtin_amdgcn_s_wait_tensorcnt(0);
  }
  __syncthreads();
#else
  for (int i = tid; i < 16 * Hh; i += 128) xkb[i >> 7][i & 127] = xg[i];
  __syncthreads();
#endif
  TFW ts[2]; ts[0] = t0; ts[1] = t1;
  for (int tl = 0; tl < 2; ++tl) {
    const TFW T = ts[tl];
    // LN1 -> ab (f16)
    if (tid < 16) {
      float s = 0.f, ss = 0.f;
      for (int c = 0; c < Hh; ++c) { const float v = xkb[tid][c]; s += v; ss += v * v; }
      const float m = s * (1.0f / Hh), var = ss * (1.0f / Hh) - m * m;
      const float r = rsqrtf(var + 1e-5f);
      for (int c = 0; c < Hh; ++c)
        ab[tid][c] = (_Float16)((xkb[tid][c] - m) * r * T.l1g[c] + T.l1b[c]);
    }
    __syncthreads();
    // qkv projection: 24 N-tiles across 4 waves
    for (int j = 0; j < 6; ++j) {
      const int nt = w + 4 * j;
      v8f acc = vzero8();
      for (int kt = 0; kt < 4; ++kt) {
        v16h a = load_a16(&ab[0][0] + kt * 32, Hh, lane);
        acc = wmma16(a, load_bp(T.ipP, kt * 24 + nt, lane), acc);
      }
      const float bias = T.ipB[nt * 16 + n];
#pragma unroll
      for (int i = 0; i < 8; ++i)
        qkvb[i + 8 * hi][nt * 16 + n] = (_Float16)(acc[i] + bias);
    }
    __syncthreads();
    // attention: one head per wave
    {
      const int hh = w;
      v16h aq = load_a16(&qkvb[0][hh * 32], 384, lane);
      v16h bk;
      {
        const int koff = (lane & 16) ? 16 : 0;
        const _Float16* p = &qkvb[n][128 + hh * 32 + koff];
#pragma unroll
        for (int e = 0; e < 16; ++e) bk[e] = p[e];
      }
      v8f sacc = wmma16(aq, bk, vzero8());
      const float qs = 0.17677669529663687f;  // 1/sqrt(32)
#pragma unroll
      for (int i = 0; i < 8; ++i) scb[hh][i + 8 * hi][n] = sacc[i] * qs;
      if (lane < 16) {
        float mx = -1e30f;
#pragma unroll
        for (int c = 0; c < 16; ++c) mx = fmaxf(mx, scb[hh][lane][c]);
        float s = 0.f, e2[16];
#pragma unroll
        for (int c = 0; c < 16; ++c) { e2[c] = __expf(scb[hh][lane][c] - mx); s += e2[c]; }
        const float inv = 1.0f / s;
#pragma unroll
        for (int c = 0; c < 16; ++c) pb[hh][lane][c] = (_Float16)(e2[c] * inv);
#pragma unroll
        for (int c = 16; c < 32; ++c) pb[hh][lane][c] = (_Float16)0.f;
      }
      for (int nt2 = 0; nt2 < 2; ++nt2) {
        v16h ap = load_a16(&pb[hh][0][0], 32, lane);
        v16h bv;
        {
          const int koff = (lane & 16) ? 16 : 0;
#pragma unroll
          for (int e = 0; e < 16; ++e)
            bv[e] = qkvb[koff + e][256 + hh * 32 + nt2 * 16 + n];
        }
        v8f oacc = wmma16(ap, bv, vzero8());
#pragma unroll
        for (int i = 0; i < 8; ++i)
          ob[i + 8 * hi][hh * 32 + nt2 * 16 + n] = (_Float16)oacc[i];
      }
    }
    __syncthreads();
    // out_proj + residual into xkb
    for (int j = 0; j < 2; ++j) {
      const int nt = w + 4 * j;
      v8f acc = vzero8();
      for (int kt = 0; kt < 4; ++kt) {
        v16h a = load_a16(&ob[0][0] + kt * 32, Hh, lane);
        acc = wmma16(a, load_bp(T.opP, kt * 8 + nt, lane), acc);
      }
      const float bias = T.opB[nt * 16 + n];
#pragma unroll
      for (int i = 0; i < 8; ++i) xkb[i + 8 * hi][nt * 16 + n] += acc[i] + bias;
    }
    __syncthreads();
    // LN2 -> ab
    if (tid < 16) {
      float s = 0.f, ss = 0.f;
      for (int c = 0; c < Hh; ++c) { const float v = xkb[tid][c]; s += v; ss += v * v; }
      const float m = s * (1.0f / Hh), var = ss * (1.0f / Hh) - m * m;
      const float r = rsqrtf(var + 1e-5f);
      for (int c = 0; c < Hh; ++c)
        ab[tid][c] = (_Float16)((xkb[tid][c] - m) * r * T.l2g[c] + T.l2b[c]);
    }
    __syncthreads();
    // ff1 (128 -> 256, relu)
    for (int j = 0; j < 4; ++j) {
      const int nt = w + 4 * j;
      v8f acc = vzero8();
      for (int kt = 0; kt < 4; ++kt) {
        v16h a = load_a16(&ab[0][0] + kt * 32, Hh, lane);
        acc = wmma16(a, load_bp(T.f1P, kt * 16 + nt, lane), acc);
      }
      const float bias = T.f1B[nt * 16 + n];
#pragma unroll
      for (int i = 0; i < 8; ++i) {
        float v = acc[i] + bias;
        hb[i + 8 * hi][nt * 16 + n] = (_Float16)(v > 0.f ? v : 0.f);
      }
    }
    __syncthreads();
    // ff2 (256 -> 128) + residual
    for (int j = 0; j < 2; ++j) {
      const int nt = w + 4 * j;
      v8f acc = vzero8();
      for (int kt = 0; kt < 8; ++kt) {
        v16h a = load_a16(&hb[0][0] + kt * 32, 256, lane);
        acc = wmma16(a, load_bp(T.f2P, kt * 8 + nt, lane), acc);
      }
      const float bias = T.f2B[nt * 16 + n];
#pragma unroll
      for (int i = 0; i < 8; ++i) xkb[i + 8 * hi][nt * 16 + n] += acc[i] + bias;
    }
    __syncthreads();
  }
  for (int i = tid; i < 16 * Hh; i += 128) xg[i] = xkb[i >> 7][i & 127];
}

// ---------------------------------------------------------------------------
// Kernel 9: final head  out = -dist_sq*base_scale + alpha * (xk @ headW^T + b)
// ---------------------------------------------------------------------------
__global__ __launch_bounds__(256) void k_final(
    const float* __restrict__ sh, const float* __restrict__ xk,
    const float* __restrict__ hW, const float* __restrict__ hB,
    const float* __restrict__ base_scale, const float* __restrict__ alpha,
    float* __restrict__ out) {
  const int idx = blockIdx.x * 256 + threadIdx.x;
  if (idx >= Bq * Kq) return;
  const int b = idx >> 4, k = idx & 15;
  float m0 = 0.f, m1 = 0.f;
  for (int l = 0; l < Lq; ++l) {
    const size_t s = ((size_t)(b * Lq + l) * Kq + k) * 2;
    m0 += sh[s]; m1 += sh[s + 1];
  }
  m0 *= (1.0f / Lq); m1 *= (1.0f / Lq);
  const float iv = 0.7071067811865475f;
  const float cx[4] = {iv, -iv, -iv, iv};
  const float cy[4] = {iv, iv, -iv, -iv};
  const float* xr = xk + (size_t)idx * Hh;
  const float bs = base_scale[0], al = alpha[0];
#pragma unroll
  for (int c = 0; c < 4; ++c) {
    const float d0 = m0 - cx[c], d1 = m1 - cy[c];
    float hv = hB[c];
    for (int ch = 0; ch < Hh; ++ch) hv += xr[ch] * hW[c * Hh + ch];
    out[(size_t)idx * 4 + c] = -(d0 * d0 + d1 * d1) * bs + al * hv;
  }
}

__global__ void k_zero(float* __restrict__ p, int nEl) {
  const int i = blockIdx.x * 256 + threadIdx.x;
  if (i < nEl) p[i] = 0.f;
}

// ---------------------------------------------------------------------------
// Host launcher. Params flattened in jax pytree (sorted-key) order.
// ---------------------------------------------------------------------------
extern "C" void kernel_launch(void* const* d_in, const int* in_sizes, int n_in,
                              void* d_out, int out_size, void* d_ws, size_t ws_size,
                              hipStream_t stream) {
  (void)in_sizes; (void)n_in; (void)out_size; (void)ws_size;
  const float* sh  = (const float*)d_in[0];
  const float* Hv  = (const float*)d_in[1];
  const float* yv  = (const float*)d_in[2];
  const float* bwf = (const float*)d_in[3];
  const float* snr = (const float*)d_in[4];
#define PP(i) ((const float*)d_in[(i)])
  const float* alpha = PP(5);
  const float* apW0 = PP(6);  const float* apB0 = PP(7);
  const float* apW1 = PP(8);  const float* apB1 = PP(9);
  const float* base_scale = PP(10);
  const float* bnb = PP(11);  const float* bng = PP(12);
  const float* cW0 = PP(13);  const float* cB0 = PP(14);
  const float* cW1 = PP(15);  const float* cB1 = PP(16);
  const float* dW0 = PP(17);  const float* dB0 = PP(18);
  const float* dW1 = PP(19);  const float* dB1 = PP(20);
  const float* fW0 = PP(21);  const float* fB0 = PP(22);
  const float* fW1 = PP(23);  const float* fB1 = PP(24);
  const float* hdW = PP(55);  const float* hdB = PP(56);
  const float* mW0 = PP(57);  const float* mB0 = PP(58);
  const float* mW1 = PP(59);  const float* mB1 = PP(60);

  uint8_t* ws = (uint8_t*)d_ws;
  size_t off = 0;
  auto carve = [&](size_t bytes) -> void* {
    void* p = ws + off;
    off = (off + bytes + 255) & ~(size_t)255;
    return p;
  };
  _Float16* X    = (_Float16*)carve((size_t)Pn * XLD * 2);
  _Float16* node = (_Float16*)carve((size_t)Pn * Hh * 2);
  _Float16* hbuf = (_Float16*)carve((size_t)Pn * Hh * 2);
  float* mrc     = (float*)carve((size_t)Bq * Kq * Hh * 4);
  float* msum    = (float*)carve((size_t)Bq * Kq * Hh * 4);
  float* bnsum   = (float*)carve(512);
  float* bnss    = (float*)carve(512);
  float* scoresb = (float*)carve((size_t)Pn * 4);
  float* xkb     = (float*)carve((size_t)Bq * Kq * Hh * 4);
  _Float16* fW1p = (_Float16*)carve(3 * 8 * 1024);
  _Float16* fW2p = (_Float16*)carve(4 * 8 * 1024);
  _Float16 *mW1p[3], *mW2p[3], *uW1p[3], *uW2p[3];
  for (int i = 0; i < 3; ++i) {
    mW1p[i] = (_Float16*)carve(4 * 8 * 1024);
    mW2p[i] = (_Float16*)carve(4 * 8 * 1024);
    uW1p[i] = (_Float16*)carve(8 * 8 * 1024);
    uW2p[i] = (_Float16*)carve(4 * 8 * 1024);
  }
  _Float16* apP = (_Float16*)carve(4 * 4 * 1024);
  _Float16 *ipP[2], *opP[2], *f1P[2], *f2P[2];
  for (int t = 0; t < 2; ++t) {
    ipP[t] = (_Float16*)carve(4 * 24 * 1024);
    opP[t] = (_Float16*)carve(4 * 8 * 1024);
    f1P[t] = (_Float16*)carve(4 * 16 * 1024);
    f2P[t] = (_Float16*)carve(8 * 8 * 1024);
  }

  auto pack = [&](const float* W, int ldw, int kmax, int kt, int nt,
                  _Float16* dst) {
    const int total = kt * nt * 32;
    k_pack<<<(total + 255) / 256, 256, 0, stream>>>(W, ldw, kmax, kt, nt, dst);
  };
  pack(fW0, 70, 70, 3, 8, fW1p);
  pack(fW1, 128, 128, 4, 8, fW2p);
  for (int i = 0; i < 3; ++i) {
    const int g = 25 + 10 * i;
    pack(PP(g + 2), 128, 128, 4, 8, mW1p[i]);
    pack(PP(g + 4), 128, 128, 4, 8, mW2p[i]);
    pack(PP(g + 6), 256, 256, 8, 8, uW1p[i]);
    pack(PP(g + 8), 128, 128, 4, 8, uW2p[i]);
  }
  pack(apW0, 128, 128, 4, 4, apP);
  for (int t = 0; t < 2; ++t) {
    const int a = 61 + 12 * t;
    pack(PP(a + 4), 128, 128, 4, 24, ipP[t]);  // in_proj
    pack(PP(a + 10), 128, 128, 4, 8, opP[t]);  // out_proj
    pack(PP(a + 0), 128, 128, 4, 16, f1P[t]);  // ff1
    pack(PP(a + 2), 256, 256, 8, 8, f2P[t]);   // ff2
  }

  k_zero<<<1, 256, 0, stream>>>(bnsum, 128);
  k_zero<<<1, 256, 0, stream>>>(bnss, 128);
  k_mrc<<<Bq, 256, 0, stream>>>(Hv, yv, mW0, mB0, mW1, mB1, mrc);
  k_feat<<<Bq * Lq, 128, 0, stream>>>(sh, Hv, bwf, snr, dW0, dB0, dW1, dB1,
                                      cW0, cB0, cW1, cB1, X);
  k_fusion<<<Pn / 64, 128, 0, stream>>>(X, fW1p, fB0, fW2p, fB1, node);
  k_bnstat<<<512, 256, 0, stream>>>(node, bnsum, bnss);
  k_bnapply<<<16384, 256, 0, stream>>>(node, bnsum, bnss, bng, bnb, hbuf);

  for (int i = 0; i < 3; ++i) {
    const int g = 25 + 10 * i;
    const float* lnb_ = PP(g + 0); const float* lng_ = PP(g + 1);
    const float* gmB0 = PP(g + 3); const float* gmB1 = PP(g + 5);
    const float* guB0 = PP(g + 7); const float* guB1 = PP(g + 9);
    k_zero<<<512, 256, 0, stream>>>(msum, Bq * Kq * Hh);
    k_msg<<<Pn / 64, 128, 0, stream>>>(hbuf, mW1p[i], gmB0, mW2p[i], gmB1, msum);
    k_upd<<<Pn / 64, 128, 0, stream>>>(hbuf, msum, uW1p[i], guB0, uW2p[i], guB1,
                                       lng_, lnb_);
  }

  k_score<<<Pn / 64, 128, 0, stream>>>(hbuf, apP, apB0, apW1, apB1, scoresb);
  k_pool<<<Bq * Kq, 128, 0, stream>>>(hbuf, scoresb, mrc, xkb);

  TFW t0, t1;
  {
    const int a = 61;
    t0.f1P = f1P[0]; t0.f1B = PP(a + 1); t0.f2P = f2P[0]; t0.f2B = PP(a + 3);
    t0.ipP = ipP[0]; t0.ipB = PP(a + 5);
    t0.l1b = PP(a + 6); t0.l1g = PP(a + 7); t0.l2b = PP(a + 8); t0.l2g = PP(a + 9);
    t0.opP = opP[0]; t0.opB = PP(a + 11);
  }
  {
    const int a = 73;
    t1.f1P = f1P[1]; t1.f1B = PP(a + 1); t1.f2P = f2P[1]; t1.f2B = PP(a + 3);
    t1.ipP = ipP[1]; t1.ipB = PP(a + 5);
    t1.l1b = PP(a + 6); t1.l1g = PP(a + 7); t1.l2b = PP(a + 8); t1.l2g = PP(a + 9);
    t1.opP = opP[1]; t1.opB = PP(a + 11);
  }
  k_tf<<<Bq, 128, 0, stream>>>(xkb, t0, t1);
  k_final<<<(Bq * Kq + 255) / 256, 256, 0, stream>>>(sh, xkb, hdW, hdB,
                                                     base_scale, alpha,
                                                     (float*)d_out);
#undef PP
}
